// AVEQA_66391604461839
// MI455X (gfx1250) — matrix operand order repaired and checked by reading, not compiled, not run on hardware
//
#include <hip/hip_runtime.h>

// ---------------------------------------------------------------------------
// AVEQA (dual-BERT QA) forward for MI455X / gfx1250, wave32, WMMA bf16 path.
//   B=16, S=128, H=768, L=12, NH=12, DK=64, FF=3072, V=30522, SPAN=4
// GEMMs: v_wmma_f32_16x16x32_bf16, weights pre-converted to bf16 per layer,
// double-buffered LDS staging, guaranteed ds_load_b128 fragment reads.
// ---------------------------------------------------------------------------

#define Bb   16
#define Ss   128
#define Hh   768
#define Ll   12
#define NHh  12
#define DKk  64
#define FFf  3072
#define Vv   30522
#define BSs  (Bb * Ss)          // 2048
#define NEGC 1000000000.0f

// output layout (flat floats, return-order concat)
#define OUT_BEGIN 0
#define OUT_END   2048
#define OUT_NOANS 2560
#define OUT_BGT   2592
#define OUT_CPD   (2592 + 64 * Vv)   // 1956000

typedef __attribute__((ext_vector_type(16))) __bf16       v16bf;
typedef __attribute__((ext_vector_type(8)))  float        v8f;
typedef __attribute__((ext_vector_type(4)))  float        f32x4;
typedef __attribute__((ext_vector_type(4)))  unsigned int u32x4;

union Frag { v16bf bf; unsigned int u[8]; };
union BF4  { __bf16 h[4]; unsigned long long u64; };
union BF8  { __bf16 h[8]; u32x4 u4; };

__device__ __forceinline__ f32x4 ldg_f4(const float* p) {
    f32x4 v; __builtin_memcpy(&v, p, 16); return v;   // -> global_load_b128
}
__device__ __forceinline__ u32x4 ldg_u4(const __bf16* p) {
    u32x4 v; __builtin_memcpy(&v, p, 16); return v;   // -> global_load_b128
}
// Two aligned 16B LDS chunks -> one WMMA operand (emits 2x ds_load_b128)
__device__ __forceinline__ void ldsFrag(Frag& f, const __bf16* p0, const __bf16* p1) {
    u32x4 lo = *(const u32x4*)p0;
    u32x4 hi = *(const u32x4*)p1;
    f.u[0] = lo.x; f.u[1] = lo.y; f.u[2] = lo.z; f.u[3] = lo.w;
    f.u[4] = hi.x; f.u[5] = hi.y; f.u[6] = hi.z; f.u[7] = hi.w;
}

// LDS strides (elements): multiples of 8 elems (16B) to keep frag alignment
#define AS_STR 72     // rows of 64 bf16, padded
#define VT_STR 136    // rows of 128 bf16, padded
#define SC_STR 129    // scores rows of 128 f32, padded

// ---------------------------------------------------------------------------
// bf16-WMMA GEMM:  C[M,N] = epi(A[M,K](f32) @ W[K,N](bf16) + bias [+resid])
// 128x64 tile / block, 256 threads (8 waves), K tiled by 64 (2 wmma k-steps),
// double-buffered LDS; loads issued before compute, stores after (latency hiding).
// Requires: M % 128 == 0, K % 64 == 0.  N arbitrary (tail block scalar path).
// ---------------------------------------------------------------------------
template <int ACT, bool RESID>
__global__ __launch_bounds__(256) void gemm_wmma_k(
    const float* __restrict__ A, const __bf16* __restrict__ W,
    const float* __restrict__ bias, const float* __restrict__ resid,
    float* __restrict__ C, int M, int N, int K)
{
    __shared__ __align__(16) __bf16 As[2][128][AS_STR];   // [m][k]
    __shared__ __align__(16) __bf16 Bs[2][64][AS_STR];    // transposed: [n][k]

    const int tid  = threadIdx.x;
    const int wave = tid >> 5;
    const int lane = tid & 31;
    const int r    = lane & 15;
    const int hh   = lane >> 4;
    const int rowBase = blockIdx.y * 128;
    const int colBase = blockIdx.x * 64;
    const int m0 = wave * 16;
    const bool fullN = (colBase + 64 <= N);

    v8f acc[4] = {};
    f32x4 ra[8];
    u32x4 rb[2];

    auto loadTiles = [&](int k0) {
#pragma unroll
        for (int i = 0; i < 8; ++i) {
            int idx = i * 256 + tid;           // 2048 f32x4 chunks (128x64 f32)
            int mm = idx >> 4, c4 = (idx & 15) * 4;
            ra[i] = ldg_f4(&A[(size_t)(rowBase + mm) * K + k0 + c4]);
        }
        if (fullN) {
#pragma unroll
            for (int i = 0; i < 2; ++i) {
                int idx = i * 256 + tid;       // 512 chunks of 8 bf16 (64x64)
                int kk = idx >> 3, c8 = (idx & 7) * 8;
                rb[i] = ldg_u4(&W[(size_t)(k0 + kk) * N + colBase + c8]);
            }
        }
    };
    auto storeTiles = [&](int p, int k0) {
#pragma unroll
        for (int i = 0; i < 8; ++i) {
            int idx = i * 256 + tid;
            int mm = idx >> 4, c4 = (idx & 15) * 4;
            BF4 t;
            t.h[0] = (__bf16)ra[i].x; t.h[1] = (__bf16)ra[i].y;
            t.h[2] = (__bf16)ra[i].z; t.h[3] = (__bf16)ra[i].w;
            *(unsigned long long*)&As[p][mm][c4] = t.u64;       // ds_store_b64
        }
        if (fullN) {
#pragma unroll
            for (int i = 0; i < 2; ++i) {
                int idx = i * 256 + tid;
                int kk = idx >> 3, c8 = (idx & 7) * 8;
                BF8 t; t.u4 = rb[i];
#pragma unroll
                for (int j = 0; j < 8; ++j) Bs[p][c8 + j][kk] = t.h[j];
            }
        } else {   // vocab tail block - scalar guarded (rare)
            for (int i = 0; i < 16; ++i) {
                int idx = i * 256 + tid;
                int kk = idx >> 6, nn = idx & 63;
                int gn = colBase + nn;
                Bs[p][nn][kk] = (gn < N) ? W[(size_t)(k0 + kk) * N + gn] : (__bf16)0.0f;
            }
        }
    };
    auto compute = [&](int p) {
#pragma unroll
        for (int ks = 0; ks < 2; ++ks) {
            const int kb = ks * 32;
            Frag a;
            const __bf16* pa = &As[p][m0 + r][kb + hh * 8];
            ldsFrag(a, pa, pa + 16);
#pragma unroll
            for (int c = 0; c < 4; ++c) {
                Frag b;
                const __bf16* pb = &Bs[p][c * 16 + r][kb + hh * 8];
                ldsFrag(b, pb, pb + 16);
                acc[c] = __builtin_amdgcn_wmma_f32_16x16x32_bf16(
                    false, a.bf, false, b.bf, (short)0, acc[c], false, false);
            }
        }
    };

    const int nk = K >> 6;
    loadTiles(0);
    storeTiles(0, 0);
    __syncthreads();
    for (int t = 0; t < nk; ++t) {
        const int p = t & 1;
        if (t + 1 < nk) loadTiles((t + 1) << 6);   // issue next-tile loads first
        compute(p);                                 // WMMA hides load latency
        if (t + 1 < nk) storeTiles(p ^ 1, (t + 1) << 6);
        __syncthreads();
    }

    // C layout: lane (hh,r) holds n = r, rows m = hh*8 + vgpr
#pragma unroll
    for (int c = 0; c < 4; ++c) {
        int n = colBase + c * 16 + r;
        if (n >= N) continue;
        float bv = bias[n];
#pragma unroll
        for (int rr = 0; rr < 8; ++rr) {
            int m = rowBase + m0 + hh * 8 + rr;
            float v = acc[c][rr] + bv;
            if (RESID) v += resid[(size_t)m * N + n];
            if (ACT == 1) {
                float t = 0.7978845608028654f * (v + 0.044715f * v * v * v);
                v = 0.5f * v * (1.0f + tanhf(t));
            }
            C[(size_t)m * N + n] = v;
        }
    }
}

// ---------------------------------------------------------------------------
// f32 -> bf16 bulk converter (weights); n % 4 == 0
// ---------------------------------------------------------------------------
__global__ __launch_bounds__(256) void cvt_bf16_k(
    const float* __restrict__ src, __bf16* __restrict__ dst, long n)
{
    long i = ((long)blockIdx.x * 256 + (long)threadIdx.x) * 4;
    if (i + 4 <= n) {
        f32x4 v = ldg_f4(&src[i]);
        BF4 t;
        t.h[0] = (__bf16)v.x; t.h[1] = (__bf16)v.y;
        t.h[2] = (__bf16)v.z; t.h[3] = (__bf16)v.w;
        *(unsigned long long*)&dst[i] = t.u64;
    }
}

// ---------------------------------------------------------------------------
// Fused attention per (batch, head): scores = QK^T/8 + mask, softmax, P@V.
// qkv: [B,S,3H]; out: [B,S,H]. One 128-thread block per (b,h); dynamic LDS.
// ---------------------------------------------------------------------------
__global__ __launch_bounds__(128) void attn_k(
    const float* __restrict__ qkv, const int* __restrict__ mask,
    float* __restrict__ out)
{
    extern __shared__ char smem[];
    __bf16* Qb = (__bf16*)smem;                   // [128][AS_STR]
    __bf16* Kb = Qb + 128 * AS_STR;               // [128][AS_STR]
    __bf16* Vt = Kb + 128 * AS_STR;               // [64][VT_STR]  V transposed
    float*  Sc = (float*)(Vt + 64 * VT_STR);      // [128][SC_STR]
    __bf16* Pb = (__bf16*)(Sc + 128 * SC_STR);    // [128][VT_STR]

    const int b = blockIdx.x / NHh, h = blockIdx.x % NHh;
    const int tid = threadIdx.x, wave = tid >> 5, lane = tid & 31;
    const int r = lane & 15, hh = lane >> 4;
    const size_t rowStride = 3 * Hh;
    const float* base = qkv + (size_t)(b * Ss) * rowStride + h * DKk;

    // stage Q/K/V -> bf16 LDS (b128 global loads, b64 LDS stores for Q/K)
#pragma unroll 4
    for (int i = 0; i < 16; ++i) {
        int idx = i * 128 + tid;         // 2048 f32x4 chunks per matrix
        int s = idx >> 4, d4 = (idx & 15) * 4;
        const float* p = base + (size_t)s * rowStride;
        f32x4 q = ldg_f4(&p[d4]);
        f32x4 k = ldg_f4(&p[Hh + d4]);
        f32x4 v = ldg_f4(&p[2 * Hh + d4]);
        BF4 tq, tk;
        tq.h[0] = (__bf16)q.x; tq.h[1] = (__bf16)q.y; tq.h[2] = (__bf16)q.z; tq.h[3] = (__bf16)q.w;
        tk.h[0] = (__bf16)k.x; tk.h[1] = (__bf16)k.y; tk.h[2] = (__bf16)k.z; tk.h[3] = (__bf16)k.w;
        *(unsigned long long*)&Qb[s * AS_STR + d4] = tq.u64;
        *(unsigned long long*)&Kb[s * AS_STR + d4] = tk.u64;
        Vt[(d4 + 0) * VT_STR + s] = (__bf16)v.x;
        Vt[(d4 + 1) * VT_STR + s] = (__bf16)v.y;
        Vt[(d4 + 2) * VT_STR + s] = (__bf16)v.z;
        Vt[(d4 + 3) * VT_STR + s] = (__bf16)v.w;
    }
    __syncthreads();

    const int m0 = wave * 32;

    // scores = Q @ K^T  (rows m0..m0+31 per wave, 8 col tiles, DK=64 in 2 steps)
    for (int ct = 0; ct < 8; ++ct) {
        v8f a0 = {}, a1 = {};
#pragma unroll
        for (int ks = 0; ks < 2; ++ks) {
            int d0 = ks * 32;
            Frag fa, fb, fk;
            const __bf16* pa = &Qb[(m0 + r) * AS_STR + d0 + hh * 8];
            const __bf16* pb = &Qb[(m0 + 16 + r) * AS_STR + d0 + hh * 8];
            const __bf16* pk = &Kb[(ct * 16 + r) * AS_STR + d0 + hh * 8];
            ldsFrag(fa, pa, pa + 16);
            ldsFrag(fb, pb, pb + 16);
            ldsFrag(fk, pk, pk + 16);
            a0 = __builtin_amdgcn_wmma_f32_16x16x32_bf16(false, fa.bf, false, fk.bf, (short)0, a0, false, false);
            a1 = __builtin_amdgcn_wmma_f32_16x16x32_bf16(false, fb.bf, false, fk.bf, (short)0, a1, false, false);
        }
#pragma unroll
        for (int rr = 0; rr < 8; ++rr) {
            Sc[(m0 + hh * 8 + rr) * SC_STR + ct * 16 + r]      = a0[rr];
            Sc[(m0 + 16 + hh * 8 + rr) * SC_STR + ct * 16 + r] = a1[rr];
        }
    }
    __syncthreads();

    // masked softmax, one row per thread
    {
        const int m = tid;
        float mx = -1e30f;
        for (int n = 0; n < Ss; ++n) {
            float v = Sc[m * SC_STR + n] * 0.125f
                    + (float)(mask[b * Ss + n] - 1) * NEGC;
            Sc[m * SC_STR + n] = v;
            mx = fmaxf(mx, v);
        }
        float sum = 0.0f;
        for (int n = 0; n < Ss; ++n) {
            float e = __expf(Sc[m * SC_STR + n] - mx);
            Sc[m * SC_STR + n] = e;
            sum += e;
        }
        float inv = 1.0f / sum;
        for (int n = 0; n < Ss; ++n)
            Pb[m * VT_STR + n] = (__bf16)(Sc[m * SC_STR + n] * inv);
    }
    __syncthreads();

    // out = P @ V  (4 col tiles of DK, S=128 contraction in 4 steps)
    for (int ct = 0; ct < 4; ++ct) {
        v8f a0 = {}, a1 = {};
#pragma unroll
        for (int ks = 0; ks < 4; ++ks) {
            int k0 = ks * 32;
            Frag fa, fb, fv;
            const __bf16* pa = &Pb[(m0 + r) * VT_STR + k0 + hh * 8];
            const __bf16* pb = &Pb[(m0 + 16 + r) * VT_STR + k0 + hh * 8];
            const __bf16* pv = &Vt[(ct * 16 + r) * VT_STR + k0 + hh * 8];
            ldsFrag(fa, pa, pa + 16);
            ldsFrag(fb, pb, pb + 16);
            ldsFrag(fv, pv, pv + 16);
            a0 = __builtin_amdgcn_wmma_f32_16x16x32_bf16(false, fa.bf, false, fv.bf, (short)0, a0, false, false);
            a1 = __builtin_amdgcn_wmma_f32_16x16x32_bf16(false, fb.bf, false, fv.bf, (short)0, a1, false, false);
        }
#pragma unroll
        for (int rr = 0; rr < 8; ++rr) {
            out[(size_t)(b * Ss + m0 + hh * 8 + rr) * Hh + h * DKk + ct * 16 + r]      = a0[rr];
            out[(size_t)(b * Ss + m0 + 16 + hh * 8 + rr) * Hh + h * DKk + ct * 16 + r] = a1[rr];
        }
    }
}

// ---------------------------------------------------------------------------
// Small VALU kernels
// ---------------------------------------------------------------------------
__device__ __forceinline__ float blockReduceSum256(float v, float* red) {
    int t = threadIdx.x;
    red[t] = v; __syncthreads();
    for (int st = 128; st > 0; st >>= 1) {
        if (t < st) red[t] += red[t + st];
        __syncthreads();
    }
    float r = red[0]; __syncthreads();
    return r;
}
__device__ __forceinline__ float blockReduceMax256(float v, float* red) {
    int t = threadIdx.x;
    red[t] = v; __syncthreads();
    for (int st = 128; st > 0; st >>= 1) {
        if (t < st) red[t] = fmaxf(red[t], red[t + st]);
        __syncthreads();
    }
    float r = red[0]; __syncthreads();
    return r;
}

__global__ __launch_bounds__(256) void layernorm_k(
    const float* __restrict__ in, const float* __restrict__ g,
    const float* __restrict__ bt, float* __restrict__ out)
{
    __shared__ float red[256];
    const int row = blockIdx.x;
    const float* x = in + (size_t)row * Hh;
    float v0[3], s = 0.0f;
#pragma unroll
    for (int j = 0; j < 3; ++j) { v0[j] = x[threadIdx.x + j * 256]; s += v0[j]; }
    float mu = blockReduceSum256(s, red) * (1.0f / Hh);
    float s2 = 0.0f;
#pragma unroll
    for (int j = 0; j < 3; ++j) { float d = v0[j] - mu; s2 += d * d; }
    float var = blockReduceSum256(s2, red) * (1.0f / Hh);
    float rs = rsqrtf(var + 1e-12f);
#pragma unroll
    for (int j = 0; j < 3; ++j) {
        int i = threadIdx.x + j * 256;
        out[(size_t)row * Hh + i] = (v0[j] - mu) * rs * g[i] + bt[i];
    }
}

__global__ __launch_bounds__(256) void embed_ctx_k(
    const int* __restrict__ ids, const float* __restrict__ tok,
    const float* __restrict__ pos, const float* __restrict__ seg,
    float* __restrict__ out)
{
    const int row = blockIdx.x;
    const int s = row & (Ss - 1);
    const int id = ids[row];
#pragma unroll
    for (int j = 0; j < 3; ++j) {
        int i = threadIdx.x + j * 256;
        out[(size_t)row * Hh + i] =
            tok[(size_t)id * Hh + i] + pos[(size_t)s * Hh + i] + seg[Hh + i]; // seg_emb[1]
    }
}

__global__ __launch_bounds__(256) void embed_bert_k(
    const int* __restrict__ ids, const int* __restrict__ tt,
    const float* __restrict__ tok, const float* __restrict__ pos,
    const float* __restrict__ type_emb, float* __restrict__ out)
{
    const int row = blockIdx.x;
    const int s = row & (Ss - 1);
    const int id = ids[row];
    const int t = tt[row];
#pragma unroll
    for (int j = 0; j < 3; ++j) {
        int i = threadIdx.x + j * 256;
        out[(size_t)row * Hh + i] =
            tok[(size_t)id * Hh + i] + pos[(size_t)s * Hh + i] + type_emb[(size_t)t * Hh + i];
    }
}

// out[row] = dot(A[row,:H], w) + b0[0]
__global__ __launch_bounds__(256) void rowdot_k(
    const float* __restrict__ A, const float* __restrict__ w,
    const float* __restrict__ b0, float* __restrict__ out)
{
    __shared__ float red[256];
    const int row = blockIdx.x;
    float s = 0.0f;
    for (int i = threadIdx.x; i < Hh; i += 256)
        s += A[(size_t)row * Hh + i] * w[i];
    float r = blockReduceSum256(s, red);
    if (threadIdx.x == 0) out[row] = r + b0[0];
}

// no_answer[b, j] = dot(contextual[b,0,:], cls_w[:,j]) + cls_b[j]
__global__ __launch_bounds__(256) void cls_k(
    const float* __restrict__ ctx, const float* __restrict__ w,
    const float* __restrict__ bias, float* __restrict__ out)
{
    __shared__ float red[256];
    const int b = blockIdx.x >> 1, j = blockIdx.x & 1;
    const float* x = ctx + (size_t)(b * Ss) * Hh;
    float s = 0.0f;
    for (int i = threadIdx.x; i < Hh; i += 256)
        s += x[i] * w[i * 2 + j];
    float r = blockReduceSum256(s, red);
    if (threadIdx.x == 0) out[b * 2 + j] = r + bias[j];
}

__global__ __launch_bounds__(128) void argmax_k(
    const float* __restrict__ begin, int* __restrict__ pred)
{
    __shared__ float vs[128];
    __shared__ int   is_[128];
    const int b = blockIdx.x, t = threadIdx.x;
    vs[t] = begin[b * Ss + t]; is_[t] = t;
    __syncthreads();
    for (int st = 64; st > 0; st >>= 1) {
        if (t < st) {
            if (vs[t + st] > vs[t] || (vs[t + st] == vs[t] && is_[t + st] < is_[t])) {
                vs[t] = vs[t + st]; is_[t] = is_[t + st];
            }
        }
        __syncthreads();
    }
    if (t == 0) pred[b] = is_[0];
}

// end[b,j] = dot([h_i, h_b], we_w) + we_b
__global__ __launch_bounds__(256) void end_k(
    const float* __restrict__ ctx, const int* __restrict__ pred,
    const float* __restrict__ we_w, const float* __restrict__ we_b,
    float* __restrict__ out)
{
    __shared__ float red[256];
    const int b = blockIdx.x >> 5, j = blockIdx.x & 31;
    const int pb = pred[b];
    const int idx = min(pb + j, Ss - 1);
    const float* hi = ctx + (size_t)(b * Ss + idx) * Hh;
    const float* hb = ctx + (size_t)(b * Ss + pb) * Hh;
    float s = 0.0f;
    for (int i = threadIdx.x; i < Hh; i += 256)
        s += hi[i] * we_w[i] + hb[i] * we_w[Hh + i];
    float r = blockReduceSum256(s, red);
    if (threadIdx.x == 0) out[b * 32 + j] = r + we_b[0];
}

// gather 128 rows: [0..63] from bert_out, [64..127] from contextual
__global__ __launch_bounds__(256) void gather_k(
    const float* __restrict__ bert_out, const float* __restrict__ ctxual,
    const int* __restrict__ begin_label, float* __restrict__ gath)
{
    const int rr = blockIdx.x;
    const int half = rr >> 6, q = rr & 63;
    const int b = q >> 2, sp = q & 3;
    const int srow = b * Ss + begin_label[b] + sp;
    const float* src = (half == 0 ? bert_out : ctxual) + (size_t)srow * Hh;
    for (int i = threadIdx.x; i < Hh; i += 256)
        gath[(size_t)rr * Hh + i] = src[i];
}

// row-softmax over vocab, scatter into the two output blocks
__global__ __launch_bounds__(256) void softmaxV_k(
    const float* __restrict__ logits, float* __restrict__ out)
{
    __shared__ float red[256];
    const int r = blockIdx.x;   // 0..127
    const float* x = logits + (size_t)r * Vv;
    float mx = -1e30f;
    for (int i = threadIdx.x; i < Vv; i += 256) mx = fmaxf(mx, x[i]);
    mx = blockReduceMax256(mx, red);
    float s = 0.0f;
    for (int i = threadIdx.x; i < Vv; i += 256) s += __expf(x[i] - mx);
    s = blockReduceSum256(s, red);
    float inv = 1.0f / s;
    float* dst = out + (r < 64 ? (OUT_BGT + (size_t)r * Vv)
                               : (OUT_CPD + (size_t)(r - 64) * Vv));
    for (int i = threadIdx.x; i < Vv; i += 256)
        dst[i] = __expf(x[i] - mx) * inv;
}

// ---------------------------------------------------------------------------
// Orchestration. d_in assumed pytree-flat: 7 input arrays then params
// flattened by sorted keys (jax tree order) -- see index comments.
// ---------------------------------------------------------------------------
extern "C" void kernel_launch(void* const* d_in, const int* in_sizes, int n_in,
                              void* d_out, int out_size, void* d_ws, size_t ws_size,
                              hipStream_t stream) {
    (void)in_sizes; (void)n_in; (void)out_size; (void)ws_size;

    const int* input_ids_msk      = (const int*)d_in[0];
    const int* attention_mask_msk = (const int*)d_in[1];
    const int* input_ids          = (const int*)d_in[2];
    const int* token_type_ids     = (const int*)d_in[3];
    const int* attention_mask     = (const int*)d_in[4];
    const int* begin_label        = (const int*)d_in[5];

    const float* b_emb_ln_b = (const float*)d_in[7];
    const float* b_emb_ln_g = (const float*)d_in[8];
    const float* b_ff1_b    = (const float*)d_in[9];
    const float* b_ff1_w    = (const float*)d_in[10];
    const float* b_ff2_b    = (const float*)d_in[11];
    const float* b_ff2_w    = (const float*)d_in[12];
    const float* b_ln1_b    = (const float*)d_in[13];
    const float* b_ln1_g    = (const float*)d_in[14];
    const float* b_ln2_b    = (const float*)d_in[15];
    const float* b_ln2_g    = (const float*)d_in[16];
    const float* b_o_b      = (const float*)d_in[17];
    const float* b_o_w      = (const float*)d_in[18];
    const float* b_pos_emb  = (const float*)d_in[19];
    const float* b_qkv_b    = (const float*)d_in[20];
    const float* b_qkv_w    = (const float*)d_in[21];
    const float* b_tok_emb  = (const float*)d_in[22];
    const float* b_type_emb = (const float*)d_in[23];
    const float* cls_b      = (const float*)d_in[24];
    const float* cls_w      = (const float*)d_in[25];
    const float* c_ff1_b    = (const float*)d_in[26];
    const float* c_ff1_w    = (const float*)d_in[27];
    const float* c_ff2_b    = (const float*)d_in[28];
    const float* c_ff2_w    = (const float*)d_in[29];
    const float* c_ln1_b    = (const float*)d_in[30];
    const float* c_ln1_g    = (const float*)d_in[31];
    const float* c_ln2_b    = (const float*)d_in[32];
    const float* c_ln2_g    = (const float*)d_in[33];
    const float* c_o_b      = (const float*)d_in[34];
    const float* c_o_w      = (const float*)d_in[35];
    const float* c_qkv_b    = (const float*)d_in[36];
    const float* c_qkv_w    = (const float*)d_in[37];
    const float* c_pos_emb  = (const float*)d_in[38];
    const float* proj_b     = (const float*)d_in[39];
    const float* proj_w     = (const float*)d_in[40];
    const float* seg_emb    = (const float*)d_in[41];
    const float* c_tok_emb  = (const float*)d_in[42];
    const float* wb_b       = (const float*)d_in[43];
    const float* wb_w       = (const float*)d_in[44];
    const float* we_b       = (const float*)d_in[45];
    const float* we_w       = (const float*)d_in[46];

    float* out = (float*)d_out;

    // workspace carve-up (byte cursor, 256B aligned)
    char* wsb = (char*)d_ws;
    size_t off = 0;
    auto allocB = [&](size_t bytes) {
        void* p = wsb + off; off += (bytes + 255) & ~(size_t)255; return p;
    };
    float*  x     = (float*)allocB((size_t)BSs * Hh * 4);
    float*  t1    = (float*)allocB((size_t)BSs * Hh * 4);
    float*  qkvb  = (float*)allocB((size_t)BSs * 3 * Hh * 4);
    float*  attnb = (float*)allocB((size_t)BSs * Hh * 4);
    float*  hbuf  = (float*)allocB((size_t)BSs * Hh * 4);
    float*  ffh   = (float*)allocB((size_t)BSs * FFf * 4);
    float*  ybuf  = (float*)allocB((size_t)BSs * Hh * 4);
    float*  sbuf  = (float*)allocB((size_t)BSs * Hh * 4);
    float*  gath  = (float*)allocB((size_t)128 * Hh * 4);
    float*  plog  = (float*)allocB((size_t)128 * Vv * 4);
    int*    predb = (int*)allocB(64);
    __bf16* wqkv  = (__bf16*)allocB((size_t)Hh * 3 * Hh * 2);
    __bf16* wo    = (__bf16*)allocB((size_t)Hh * Hh * 2);
    __bf16* wff1  = (__bf16*)allocB((size_t)Hh * FFf * 2);
    __bf16* wff2  = (__bf16*)allocB((size_t)FFf * Hh * 2);
    __bf16* wproj = (__bf16*)allocB((size_t)Hh * Vv * 2);

    const size_t ATTN_LDS = (size_t)(128 * AS_STR + 128 * AS_STR + 64 * VT_STR) * 2
                          + (size_t)128 * SC_STR * 4
                          + (size_t)128 * VT_STR * 2;   // ~151.5 KB of 320 KB WGP LDS

    auto cvt = [&](const float* src, __bf16* dst, long n) {
        cvt_bf16_k<<<dim3((unsigned)((n / 4 + 255) / 256)), 256, 0, stream>>>(src, dst, n);
    };
    auto gemm = [&](const float* A, const __bf16* W, const float* bias,
                    const float* resid, float* Cc, int M, int N, int K, int act) {
        dim3 g((N + 63) / 64, (M + 127) / 128);
        if (act == 1)
            gemm_wmma_k<1, false><<<g, 256, 0, stream>>>(A, W, bias, nullptr, Cc, M, N, K);
        else if (resid)
            gemm_wmma_k<0, true><<<g, 256, 0, stream>>>(A, W, bias, resid, Cc, M, N, K);
        else
            gemm_wmma_k<0, false><<<g, 256, 0, stream>>>(A, W, bias, nullptr, Cc, M, N, K);
    };

    // ---------------- contextual (masked) encoder: pre-LN style ------------
    embed_ctx_k<<<BSs, 256, 0, stream>>>(input_ids_msk, c_tok_emb, c_pos_emb, seg_emb, x);
    for (int l = 0; l < Ll; ++l) {
        cvt(c_qkv_w + (size_t)l * Hh * 3 * Hh, wqkv, (long)Hh * 3 * Hh);
        cvt(c_o_w   + (size_t)l * Hh * Hh,     wo,   (long)Hh * Hh);
        cvt(c_ff1_w + (size_t)l * Hh * FFf,    wff1, (long)Hh * FFf);
        cvt(c_ff2_w + (size_t)l * FFf * Hh,    wff2, (long)FFf * Hh);
        layernorm_k<<<BSs, 256, 0, stream>>>(x, c_ln1_g + (size_t)l * Hh, c_ln1_b + (size_t)l * Hh, t1);
        gemm(t1, wqkv, c_qkv_b + (size_t)l * 3 * Hh, nullptr, qkvb, BSs, 3 * Hh, Hh, 0);
        attn_k<<<Bb * NHh, 128, ATTN_LDS, stream>>>(qkvb, attention_mask_msk, attnb);
        gemm(attnb, wo, c_o_b + (size_t)l * Hh, x, hbuf, BSs, Hh, Hh, 0);
        layernorm_k<<<BSs, 256, 0, stream>>>(hbuf, c_ln2_g + (size_t)l * Hh, c_ln2_b + (size_t)l * Hh, t1);
        gemm(t1, wff1, c_ff1_b + (size_t)l * FFf, nullptr, ffh, BSs, FFf, Hh, 1);
        gemm(ffh, wff2, c_ff2_b + (size_t)l * Hh, hbuf, x, BSs, Hh, FFf, 0);
    }
    // x == contextual

    // ---------------- bert encoder: post-LN style --------------------------
    embed_bert_k<<<BSs, 256, 0, stream>>>(input_ids, token_type_ids, b_tok_emb, b_pos_emb, b_type_emb, sbuf);
    layernorm_k<<<BSs, 256, 0, stream>>>(sbuf, b_emb_ln_g, b_emb_ln_b, ybuf);
    for (int l = 0; l < Ll; ++l) {
        cvt(b_qkv_w + (size_t)l * Hh * 3 * Hh, wqkv, (long)Hh * 3 * Hh);
        cvt(b_o_w   + (size_t)l * Hh * Hh,     wo,   (long)Hh * Hh);
        cvt(b_ff1_w + (size_t)l * Hh * FFf,    wff1, (long)Hh * FFf);
        cvt(b_ff2_w + (size_t)l * FFf * Hh,    wff2, (long)FFf * Hh);
        gemm(ybuf, wqkv, b_qkv_b + (size_t)l * 3 * Hh, nullptr, qkvb, BSs, 3 * Hh, Hh, 0);
        attn_k<<<Bb * NHh, 128, ATTN_LDS, stream>>>(qkvb, attention_mask, attnb);
        gemm(attnb, wo, b_o_b + (size_t)l * Hh, ybuf, sbuf, BSs, Hh, Hh, 0);
        layernorm_k<<<BSs, 256, 0, stream>>>(sbuf, b_ln1_g + (size_t)l * Hh, b_ln1_b + (size_t)l * Hh, ybuf);
        gemm(ybuf, wff1, b_ff1_b + (size_t)l * FFf, nullptr, ffh, BSs, FFf, Hh, 1);
        gemm(ffh, wff2, b_ff2_b + (size_t)l * Hh, ybuf, sbuf, BSs, Hh, FFf, 0);
        layernorm_k<<<BSs, 256, 0, stream>>>(sbuf, b_ln2_g + (size_t)l * Hh, b_ln2_b + (size_t)l * Hh, ybuf);
    }
    // ybuf == bert_out

    // ---------------- heads ------------------------------------------------
    rowdot_k<<<BSs, 256, 0, stream>>>(x, wb_w, wb_b, out + OUT_BEGIN);       // begin
    cls_k<<<Bb * 2, 256, 0, stream>>>(x, cls_w, cls_b, out + OUT_NOANS);     // no_answer
    argmax_k<<<Bb, 128, 0, stream>>>(out + OUT_BEGIN, predb);                // pred_begin
    end_k<<<Bb * 32, 256, 0, stream>>>(x, predb, we_w, we_b, out + OUT_END); // end

    gather_k<<<128, 256, 0, stream>>>(ybuf, x, begin_label, gath);           // 64 gt + 64 pred
    cvt(proj_w, wproj, (long)Hh * Vv);                                       // vocab weights
    gemm(gath, wproj, proj_b, nullptr, plog, 128, Vv, Hh, 0);                // vocab logits
    softmaxV_k<<<128, 256, 0, stream>>>(plog, out);                          // both softmax blocks
}
